// GraphTransformer_50577534878116
// MI455X (gfx1250) — compile-verified, matrix operand order checked by reading, not run on hardware
//
#include <hip/hip_runtime.h>
#include <math.h>

// ---------------- problem constants (match reference) ----------------
#define N_NODES 20000
#define N_EDGES 320000
#define IN_C    256
#define HID_C   64
#define N_HEADS 8
#define OUT_C   256

// ---------------- CDNA5 WMMA types ----------------
typedef __attribute__((ext_vector_type(16))) __bf16 v16bf;
typedef __attribute__((ext_vector_type(8)))  __bf16 v8bf;
typedef __attribute__((ext_vector_type(8)))  float  v8f;

// ---------------- small helpers ----------------
__device__ __forceinline__ unsigned short f2b(float f) {
  // fp32 -> bf16 round-to-nearest-even
  unsigned int u = __float_as_uint(f);
  u += 0x7FFFu + ((u >> 16) & 1u);
  return (unsigned short)(u >> 16);
}
__device__ __forceinline__ float b2f(unsigned short u) {
  return __uint_as_float(((unsigned int)u) << 16);
}
__device__ __forceinline__ float blo(unsigned int u) {
  return __uint_as_float(u << 16);
}
__device__ __forceinline__ float bhi(unsigned int u) {
  return __uint_as_float(u & 0xFFFF0000u);
}
__device__ __forceinline__ void atomicMaxF(float* addr, float val) {
  // monotone float-max via int/uint ordering trick (init value is -inf)
  if (val >= 0.0f) atomicMax((int*)addr, __float_as_int(val));
  else             atomicMin((unsigned int*)addr, (unsigned int)__float_as_int(val));
}

// ---------------- elementwise kernels ----------------
__global__ void f32_to_bf16_k(const float* __restrict__ in,
                              unsigned short* __restrict__ out, long long n) {
  long long t = (long long)blockIdx.x * blockDim.x + threadIdx.x;
  if (t < n) out[t] = f2b(in[t]);
}

__global__ void elu_to_bf16_k(const float* __restrict__ in,
                              unsigned short* __restrict__ out, long long n) {
  long long t = (long long)blockIdx.x * blockDim.x + threadIdx.x;
  if (t < n) {
    float x = in[t];
    float y = (x > 0.0f) ? x : (__expf(x) - 1.0f);
    out[t] = f2b(y);
  }
}

__global__ void fill_f32_k(float* __restrict__ p, float v, long long n) {
  long long t = (long long)blockIdx.x * blockDim.x + threadIdx.x;
  if (t < n) p[t] = v;
}

// Pre-pack weight W[K,M] (row-major fp32) into bf16 WMMA B-fragments.
// Tile (tk,tn) covers K rows [tk*32,+32), cols [tn*16,+16). Within a tile,
// lane l (l%16 = column, l/16 = half) holds 16 contiguous bf16 whose K index
// follows the 16-bit operand layout: j<8 -> tk*32+half*8+j ; j>=8 -> +16.
__global__ void pack_w_k(const float* __restrict__ W,
                         unsigned short* __restrict__ Wp, int K, int M) {
  long long t = (long long)blockIdx.x * blockDim.x + threadIdx.x;
  if (t >= (long long)K * M) return;
  int j    = (int)(t & 15);
  int lane = (int)((t >> 4) & 31);
  long long tile = t >> 9;
  int Nt = M >> 4;
  int tn = (int)(tile % Nt);
  int tk = (int)(tile / Nt);
  int n  = tn * 16 + (lane & 15);
  int hv = lane >> 4;
  int kk = tk * 32 + hv * 8 + ((j < 8) ? j : (16 + (j - 8)));
  Wp[t] = f2b(W[(size_t)kk * M + n]);
}

// ---------------- WMMA GEMM: C[N,M] = A[N,K](bf16) @ W[K,M](packed bf16) + bias ----
// Register blocking: each wave owns 2 m-tiles x NW n-tiles (NW = 2 or 4).
// Per k-step: 2 A-fragments + NW B-fragments feed 2*NW WMMAs.
// grid = (N/32, M/(128*NW)), block = 256 (8 waves).
template <int NW>
__global__ void gemm_bf16_k(const unsigned short* __restrict__ A,
                            const unsigned short* __restrict__ Wp,
                            const float* __restrict__ bias,
                            unsigned short* __restrict__ Cb,   // bf16 out (q/k/v) or null
                            float* __restrict__ Cf,            // f32 out (skip) or null
                            int K, int M) {
  const int lane = threadIdx.x & 31;
  const int wave = threadIdx.x >> 5;
  const int m0   = blockIdx.x * 2;                    // first of two m-tiles
  const int n0   = (blockIdx.y * 8 + wave) * NW;      // first of NW n-tiles
  const int lm   = lane & 15;
  const int hv   = lane >> 4;
  const int Nt   = M >> 4;

  const unsigned short* Arow0 = A + (size_t)(m0 * 16 + lm) * K;
  const unsigned short* Arow1 = Arow0 + (size_t)16 * K;
  const unsigned short* Bb[NW];
#pragma unroll
  for (int j = 0; j < NW; ++j)
    Bb[j] = Wp + (size_t)(n0 + j) * 512 + (size_t)lane * 16;

  v8f c0[NW], c1[NW];
#pragma unroll
  for (int j = 0; j < NW; ++j) { c0[j] = {}; c1[j] = {}; }

  const int nkt = K >> 5;
  for (int kt = 0; kt < nkt; ++kt) {
    const int abase = kt * 32 + hv * 8;
    v8bf a0lo = *(const v8bf*)(Arow0 + abase);
    v8bf a0hi = *(const v8bf*)(Arow0 + abase + 16);
    v8bf a1lo = *(const v8bf*)(Arow1 + abase);
    v8bf a1hi = *(const v8bf*)(Arow1 + abase + 16);
    v16bf a0, a1;
#pragma unroll
    for (int i = 0; i < 8; ++i) {
      a0[i] = a0lo[i]; a0[i + 8] = a0hi[i];
      a1[i] = a1lo[i]; a1[i + 8] = a1hi[i];
    }
    const size_t bo = (size_t)kt * Nt * 512;
    v16bf b[NW];
#pragma unroll
    for (int j = 0; j < NW; ++j) b[j] = *(const v16bf*)(Bb[j] + bo);
#pragma unroll
    for (int j = 0; j < NW; ++j) {
      c0[j] = __builtin_amdgcn_wmma_f32_16x16x32_bf16(false, a0, false, b[j],
                                                      (short)0, c0[j], false, false);
      c1[j] = __builtin_amdgcn_wmma_f32_16x16x32_bf16(false, a1, false, b[j],
                                                      (short)0, c1[j], false, false);
    }
  }

  const int r0 = m0 * 16 + hv * 8;   // rows for c0[*]
  const int r1 = r0 + 16;            // rows for c1[*]
#pragma unroll
  for (int j = 0; j < NW; ++j) {
    const int n  = (n0 + j) * 16 + lm;
    const float bv = bias[n];
    if (Cb) {
#pragma unroll
      for (int r = 0; r < 8; ++r) {
        Cb[(size_t)(r0 + r) * M + n] = f2b(c0[j][r] + bv);
        Cb[(size_t)(r1 + r) * M + n] = f2b(c1[j][r] + bv);
      }
    } else {
#pragma unroll
      for (int r = 0; r < 8; ++r) {
        Cf[(size_t)(r0 + r) * M + n] = c0[j][r] + bv;
        Cf[(size_t)(r1 + r) * M + n] = c1[j][r] + bv;
      }
    }
  }
}

static void launch_gemm(const unsigned short* A, const unsigned short* Wp,
                        const float* bias, unsigned short* Cb, float* Cf,
                        int K, int M, hipStream_t s) {
  if ((M % 512) == 0) {
    gemm_bf16_k<4><<<dim3(N_NODES / 32, M / 512), 256, 0, s>>>(A, Wp, bias, Cb, Cf, K, M);
  } else {
    gemm_bf16_k<2><<<dim3(N_NODES / 32, M / 256), 256, 0, s>>>(A, Wp, bias, Cb, Cf, K, M);
  }
}

// ---------------- edge-phase kernels ----------------
__global__ void edge_alpha_k(const unsigned short* __restrict__ qb,
                             const unsigned short* __restrict__ kb,
                             const int* __restrict__ src, const int* __restrict__ dst,
                             float* __restrict__ alpha, float* __restrict__ amax,
                             int H, int OC, float scale) {
  long long t = (long long)blockIdx.x * blockDim.x + threadIdx.x;
  if (t >= (long long)N_EDGES * H) return;
  int h = (int)(t % H);
  int e = (int)(t / H);
  int sN = src[e], dN = dst[e];
  // base offsets are multiples of 64 elements -> 128B aligned: vector loads OK
  const uint4* q4 = (const uint4*)(qb + (size_t)dN * H * OC + (size_t)h * OC);
  const uint4* k4 = (const uint4*)(kb + (size_t)sN * H * OC + (size_t)h * OC);
  float acc = 0.0f;
  const int nv = OC >> 3;   // 8 bf16 per uint4
  for (int i = 0; i < nv; ++i) {
    uint4 a = q4[i];
    uint4 b = k4[i];
    acc += blo(a.x) * blo(b.x) + bhi(a.x) * bhi(b.x);
    acc += blo(a.y) * blo(b.y) + bhi(a.y) * bhi(b.y);
    acc += blo(a.z) * blo(b.z) + bhi(a.z) * bhi(b.z);
    acc += blo(a.w) * blo(b.w) + bhi(a.w) * bhi(b.w);
  }
  float av = acc * scale;
  alpha[t] = av;
  atomicMaxF(&amax[(size_t)dN * H + h], av);
}

__global__ void edge_exp_k(float* __restrict__ alpha,
                           const float* __restrict__ amax,
                           float* __restrict__ asum,
                           const int* __restrict__ dst, int H) {
  long long t = (long long)blockIdx.x * blockDim.x + threadIdx.x;
  if (t >= (long long)N_EDGES * H) return;
  int h = (int)(t % H);
  int e = (int)(t / H);
  int dN = dst[e];
  float ex = __expf(alpha[t] - amax[(size_t)dN * H + h]);
  alpha[t] = ex;
  atomicAdd(&asum[(size_t)dN * H + h], ex);
}

__global__ void edge_msg_k(const unsigned short* __restrict__ vb,
                           const float* __restrict__ alpha,
                           const float* __restrict__ asum,
                           const int* __restrict__ src, const int* __restrict__ dst,
                           float* __restrict__ out, int H, int OC, int concat) {
  long long t = (long long)blockIdx.x * blockDim.x + threadIdx.x;
  long long total = (long long)N_EDGES * H * OC;
  if (t >= total) return;
  int c = (int)(t % OC);
  long long t2 = t / OC;
  int h = (int)(t2 % H);
  int e = (int)(t2 / H);
  int sN = src[e], dN = dst[e];
  float w = alpha[(size_t)e * H + h] / (asum[(size_t)dN * H + h] + 1e-16f);
  float val = w * b2f(vb[(size_t)sN * H * OC + (size_t)h * OC + c]);
  if (concat) atomicAdd(&out[(size_t)dN * H * OC + (size_t)h * OC + c], val);
  else        atomicAdd(&out[(size_t)dN * OC + c], val * (1.0f / N_HEADS));
}

// ---------------- host-side layer driver ----------------
static void run_layer(const unsigned short* hb, int Fin,
                      const float* Wq, const float* bq,
                      const float* Wk, const float* bk,
                      const float* Wv, const float* bv,
                      const float* Ws, const float* bs,
                      int OC, int concat,
                      const int* src, const int* dst,
                      unsigned short* qb, unsigned short* kb, unsigned short* vb,
                      unsigned short* wpack,
                      float* alpha, float* amax, float* asum,
                      float* outbuf, hipStream_t s) {
  const int H = N_HEADS;
  const int Fqkv  = H * OC;
  const int Fskip = concat ? H * OC : OC;

  const float* Wl[3] = { Wq, Wk, Wv };
  const float* bl[3] = { bq, bk, bv };
  unsigned short* Cl[3] = { qb, kb, vb };
  for (int i = 0; i < 3; ++i) {
    long long tw = (long long)Fin * Fqkv;
    pack_w_k<<<(unsigned)((tw + 255) / 256), 256, 0, s>>>(Wl[i], wpack, Fin, Fqkv);
    launch_gemm(hb, wpack, bl[i], Cl[i], nullptr, Fin, Fqkv, s);
  }
  { // skip projection writes the fp32 output accumulator
    long long tw = (long long)Fin * Fskip;
    pack_w_k<<<(unsigned)((tw + 255) / 256), 256, 0, s>>>(Ws, wpack, Fin, Fskip);
    launch_gemm(hb, wpack, bs, nullptr, outbuf, Fin, Fskip, s);
  }

  long long nh = (long long)N_NODES * H;
  fill_f32_k<<<(unsigned)((nh + 255) / 256), 256, 0, s>>>(amax, -INFINITY, nh);
  fill_f32_k<<<(unsigned)((nh + 255) / 256), 256, 0, s>>>(asum, 0.0f, nh);

  long long eh = (long long)N_EDGES * H;
  float scale = 1.0f / sqrtf((float)OC);
  edge_alpha_k<<<(unsigned)((eh + 255) / 256), 256, 0, s>>>(qb, kb, src, dst,
                                                           alpha, amax, H, OC, scale);
  edge_exp_k<<<(unsigned)((eh + 255) / 256), 256, 0, s>>>(alpha, amax, asum, dst, H);
  long long ehc = eh * OC;
  edge_msg_k<<<(unsigned)((ehc + 255) / 256), 256, 0, s>>>(vb, alpha, asum, src, dst,
                                                          outbuf, H, OC, concat);
}

// ---------------- entry point ----------------
extern "C" void kernel_launch(void* const* d_in, const int* in_sizes, int n_in,
                              void* d_out, int out_size, void* d_ws, size_t ws_size,
                              hipStream_t stream) {
  (void)in_sizes; (void)n_in; (void)out_size; (void)ws_size;

  const float* x = (const float*)d_in[0];
  // params flattened in insertion order per layer: Wq,bq,Wk,bk,Wv,bv,Ws,bs
  const float* P[24];
  for (int i = 0; i < 24; ++i) P[i] = (const float*)d_in[1 + i];
  const int* ei  = (const int*)d_in[25];
  const int* src = ei;
  const int* dst = ei + N_EDGES;

  char* ws = (char*)d_ws;
  size_t off = 0;
  auto carve = [&](size_t bytes) -> char* {
    size_t o = (off + 255) & ~(size_t)255;
    off = o + bytes;
    return ws + o;
  };
  unsigned short* hb    = (unsigned short*)carve((size_t)N_NODES * 512 * 2);
  unsigned short* qb    = (unsigned short*)carve((size_t)N_NODES * 2048 * 2);
  unsigned short* kb    = (unsigned short*)carve((size_t)N_NODES * 2048 * 2);
  unsigned short* vb    = (unsigned short*)carve((size_t)N_NODES * 2048 * 2);
  float*          outbf = (float*)carve((size_t)N_NODES * 512 * 4);
  float*          alpha = (float*)carve((size_t)N_EDGES * N_HEADS * 4);
  float*          amax  = (float*)carve((size_t)N_NODES * N_HEADS * 4);
  float*          asum  = (float*)carve((size_t)N_NODES * N_HEADS * 4);
  unsigned short* wpack = (unsigned short*)carve((size_t)512 * 2048 * 2);

  // x -> bf16 node features
  long long nx = (long long)N_NODES * IN_C;
  f32_to_bf16_k<<<(unsigned)((nx + 255) / 256), 256, 0, stream>>>(x, hb, nx);

  // layer 0 (concat)
  run_layer(hb, IN_C, P[0], P[1], P[2], P[3], P[4], P[5], P[6], P[7],
            HID_C, 1, src, dst, qb, kb, vb, wpack, alpha, amax, asum, outbf, stream);
  long long n0 = (long long)N_NODES * (N_HEADS * HID_C);
  elu_to_bf16_k<<<(unsigned)((n0 + 255) / 256), 256, 0, stream>>>(outbf, hb, n0);

  // layer 1 (concat)
  run_layer(hb, N_HEADS * HID_C, P[8], P[9], P[10], P[11], P[12], P[13], P[14], P[15],
            HID_C, 1, src, dst, qb, kb, vb, wpack, alpha, amax, asum, outbf, stream);
  elu_to_bf16_k<<<(unsigned)((n0 + 255) / 256), 256, 0, stream>>>(outbf, hb, n0);

  // layer 2 (mean over heads) -> d_out directly (skip GEMM overwrites, messages add)
  run_layer(hb, N_HEADS * HID_C, P[16], P[17], P[18], P[19], P[20], P[21], P[22], P[23],
            OUT_C, 0, src, dst, qb, kb, vb, wpack, alpha, amax, asum,
            (float*)d_out, stream);
}
